// SocialGraphEncoder_89026082112220
// MI455X (gfx1250) — matrix operand order repaired and checked by reading, not compiled, not run on hardware
//
#include <hip/hip_runtime.h>
#include <hip/hip_bf16.h>
#include <math.h>

typedef __attribute__((ext_vector_type(16))) __bf16 v16bf;
typedef __attribute__((ext_vector_type(8)))  __bf16 v8bf;
typedef __attribute__((ext_vector_type(8)))  float  v8f;

#define NEG_SLOPE 0.2f
#define SM_EPS    1e-16f

// ---------------------------------------------------------------------------
// split-precision packing: v = hi + lo (both bf16), fp32-accurate GEMM via
// 3 bf16 WMMAs (hi*hi + hi*lo + lo*hi)
// ---------------------------------------------------------------------------
__global__ void pack_bf16_kernel(const float* __restrict__ src,
                                 __bf16* __restrict__ hi, __bf16* __restrict__ lo,
                                 int total) {
    const int idx = blockIdx.x * blockDim.x + threadIdx.x;
    if (idx >= total) return;
    const float v = src[idx];
    const __bf16 h = (__bf16)v;
    hi[idx] = h;
    lo[idx] = (__bf16)(v - (float)h);
}

// W[K,Nout] -> transposed packed WT[Nout,K] so B-fragment loads are contiguous
__global__ void pack_bf16_t_kernel(const float* __restrict__ W,
                                   __bf16* __restrict__ hiT, __bf16* __restrict__ loT,
                                   int K, int Nout) {
    const int idx = blockIdx.x * blockDim.x + threadIdx.x;
    if (idx >= K * Nout) return;
    const int n = idx / K, k = idx % K;
    const float v = W[k * Nout + n];
    const __bf16 h = (__bf16)v;
    hiT[idx] = h;
    loT[idx] = (__bf16)(v - (float)h);
}

// ---------------------------------------------------------------------------
// GEMM: XP[M,Nout] = X[M,K] @ W[K,Nout], one wave -> 16x64 output strip.
// A frag (hi/lo) loaded once per K-step, reused across 4 N-tiles: 12 wmma/step.
// A 16x32 bf16 lane layout: lanes 0-15 K=0..15,        lanes 16-31 K=16..31
// B 32x16 lane layout:      lanes 0-15 K=0..15,        lanes 16-31 K=16..31
// ---------------------------------------------------------------------------
__global__ __launch_bounds__(256)
void gemm_wmma_split(const __bf16* __restrict__ XH, const __bf16* __restrict__ XL,
                     const __bf16* __restrict__ WHt, const __bf16* __restrict__ WLt,
                     float* __restrict__ XP, int M, int K, int Nout) {
    const int wid  = (blockIdx.x * blockDim.x + threadIdx.x) >> 5;
    const int lane = threadIdx.x & 31;
    const int ngrp   = Nout >> 6;                // groups of 4 N-tiles (64 cols)
    const int mtiles = M >> 4;
    const int tm = wid / ngrp;
    const int tg = wid % ngrp;
    if (tm >= mtiles) return;                    // wave-uniform exit (EXEC stays full)

    const int half = lane >> 4;
    const int l15  = lane & 15;
    const int arow = tm * 16 + l15;

    const __bf16* xh = XH + (size_t)arow * K;
    const __bf16* xl = XL + (size_t)arow * K;

    v8f acc[4] = {};
    for (int k0 = 0; k0 < K; k0 += 32) {
        // A fragments: two contiguous 8-elem runs per lane (16B loads)
        const v8bf r0h = *(const v8bf*)(xh + k0 + half * 16);
        const v8bf r1h = *(const v8bf*)(xh + k0 + 8 + half * 16);
        const v8bf r0l = *(const v8bf*)(xl + k0 + half * 16);
        const v8bf r1l = *(const v8bf*)(xl + k0 + 8 + half * 16);
        const v16bf ah = __builtin_shufflevector(r0h, r1h, 0,1,2,3,4,5,6,7,8,9,10,11,12,13,14,15);
        const v16bf al = __builtin_shufflevector(r0l, r1l, 0,1,2,3,4,5,6,7,8,9,10,11,12,13,14,15);
#pragma unroll
        for (int nt = 0; nt < 4; ++nt) {
            const int bcol = tg * 64 + nt * 16 + l15;
            // B fragment: 16 contiguous bf16 per lane (32B load) from transposed W
            const v16bf bh = *(const v16bf*)(WHt + (size_t)bcol * K + k0 + half * 16);
            const v16bf bl = *(const v16bf*)(WLt + (size_t)bcol * K + k0 + half * 16);
            acc[nt] = __builtin_amdgcn_wmma_f32_16x16x32_bf16(false, al, false, bh, (short)0, acc[nt], false, false);
            acc[nt] = __builtin_amdgcn_wmma_f32_16x16x32_bf16(false, ah, false, bl, (short)0, acc[nt], false, false);
            acc[nt] = __builtin_amdgcn_wmma_f32_16x16x32_bf16(false, ah, false, bh, (short)0, acc[nt], false, false);
        }
    }
#pragma unroll
    for (int nt = 0; nt < 4; ++nt) {
#pragma unroll
        for (int i = 0; i < 8; ++i) {
            const int row = tm * 16 + half * 8 + i;   // C/D: VGPR i -> M row
            XP[(size_t)row * Nout + tg * 64 + nt * 16 + l15] = acc[nt][i];
        }
    }
}

// ---------------------------------------------------------------------------
// a_s[n,h] = dot(xp[n, h*C : (h+1)*C], att_src[h]);  same for a_d
// ---------------------------------------------------------------------------
__global__ void node_att_kernel(const float* __restrict__ XP,
                                const float* __restrict__ att_src,
                                const float* __restrict__ att_dst,
                                float* __restrict__ a_s, float* __restrict__ a_d,
                                int N, int H, int C, int Nout) {
    const int idx = blockIdx.x * blockDim.x + threadIdx.x;
    if (idx >= N * H) return;
    const int n = idx / H, h = idx % H;
    const float* xr = XP + (size_t)n * Nout + h * C;
    const float* as = att_src + h * C;
    const float* ad = att_dst + h * C;
    float ss = 0.f, sd = 0.f;
    for (int c = 0; c < C; ++c) { ss += xr[c] * as[c]; sd += xr[c] * ad[c]; }
    a_s[idx] = ss;
    a_d[idx] = sd;
}

// ce[h] = dot(We[h*C:(h+1)*C], att_edge[h])  -> a_e[e,h] = ew[e]*ce[h]
__global__ void edge_coeff_kernel(const float* __restrict__ We,
                                  const float* __restrict__ att_edge,
                                  float* __restrict__ ce, int H, int C) {
    const int h = threadIdx.x;
    if (h >= H) return;
    float s = 0.f;
    for (int c = 0; c < C; ++c) s += We[h * C + c] * att_edge[h * C + c];
    ce[h] = s;
}

// order-preserving f32 <-> u32 (works for negative logits; floor encodes as 0)
__device__ __forceinline__ unsigned enc_f32(float f) {
    unsigned u = __float_as_uint(f);
    return (u & 0x80000000u) ? ~u : (u | 0x80000000u);
}
__device__ __forceinline__ float dec_u32(unsigned u) {
    return (u & 0x80000000u) ? __uint_as_float(u & 0x7FFFFFFFu) : __uint_as_float(~u);
}

// Pass A: logits + segment max over dst (atomic umax on encoded float)
__global__ void edge_logit_max_kernel(const int* __restrict__ src, const int* __restrict__ dst,
                                      const float* __restrict__ ew,
                                      const float* __restrict__ a_s, const float* __restrict__ a_d,
                                      const float* __restrict__ ce,
                                      float* __restrict__ logit, unsigned* __restrict__ mbuf,
                                      int E, int H) {
    const int idx = blockIdx.x * blockDim.x + threadIdx.x;
    if (idx >= E * H) return;
    const int e = idx / H, h = idx % H;
    const int s = src[e], d = dst[e];
    float x = a_s[s * H + h] + a_d[d * H + h] + ew[e] * ce[h];
    x = (x >= 0.f) ? x : NEG_SLOPE * x;          // leaky_relu
    logit[idx] = x;
    atomicMax(&mbuf[d * H + h], enc_f32(x));
}

// Pass B: denom[d,h] += exp(logit - m[d,h])
__global__ void edge_expsum_kernel(const int* __restrict__ dst,
                                   const float* __restrict__ logit,
                                   const unsigned* __restrict__ mbuf,
                                   float* __restrict__ den, int E, int H) {
    const int idx = blockIdx.x * blockDim.x + threadIdx.x;
    if (idx >= E * H) return;
    const int e = idx / H, h = idx % H;
    const int d = dst[e];
    atomicAdd(&den[d * H + h], __expf(logit[idx] - dec_u32(mbuf[d * H + h])));
}

// Pass B2: logit -> alpha (in place)
__global__ void edge_alpha_kernel(const int* __restrict__ dst,
                                  float* __restrict__ logit,
                                  const unsigned* __restrict__ mbuf,
                                  const float* __restrict__ den, int E, int H) {
    const int idx = blockIdx.x * blockDim.x + threadIdx.x;
    if (idx >= E * H) return;
    const int e = idx / H, h = idx % H;
    const int d = dst[e];
    const float ex = __expf(logit[idx] - dec_u32(mbuf[d * H + h]));
    logit[idx] = ex / (den[d * H + h] + SM_EPS);
}

// Pass C: out[dst, ch] += xp[src, ch] * alpha[e, ch/C]   (one thread per (e,ch))
__global__ void edge_scatter_kernel(const int* __restrict__ src, const int* __restrict__ dst,
                                    const float* __restrict__ XP,
                                    const float* __restrict__ alpha,
                                    float* __restrict__ out, int E, int Nout, int C, int H) {
    const int idx = blockIdx.x * blockDim.x + threadIdx.x;
    if (idx >= E * Nout) return;
    const int e  = idx / Nout;
    const int ch = idx % Nout;
    const int h  = ch / C;
    const float a = alpha[e * H + h];
    atomicAdd(&out[(size_t)dst[e] * Nout + ch], XP[(size_t)src[e] * Nout + ch] * a);
}

// finalize: y = (elu?) elu(out + b) : out + b
__global__ void finalize_kernel(const float* __restrict__ out, const float* __restrict__ bias,
                                float* __restrict__ y, int total, int Nout, int applyElu) {
    const int idx = blockIdx.x * blockDim.x + threadIdx.x;
    if (idx >= total) return;
    float v = out[idx] + bias[idx % Nout];
    if (applyElu) v = (v > 0.f) ? v : expm1f(v);
    y[idx] = v;
}

// ---------------------------------------------------------------------------
static void run_layer(hipStream_t stream,
                      const float* X, const float* W,
                      const float* att_src, const float* att_dst,
                      const float* We, const float* att_edge, const float* bias,
                      const int* src, const int* dst, const float* ew,
                      __bf16* XH, __bf16* XL, __bf16* WHt, __bf16* WLt,
                      float* XP, float* OUT,
                      float* a_s, float* a_d, unsigned* mbuf, float* den,
                      float* alpha, float* ce,
                      float* final_dst,
                      int N, int E, int Kin, int Nout, int H, int applyElu) {
    const int C = Nout / H;
    // pack inputs (X aliases OUT from prior layer: read-only here, dead after)
    pack_bf16_kernel<<<(N * Kin + 255) / 256, 256, 0, stream>>>(X, XH, XL, N * Kin);
    pack_bf16_t_kernel<<<(Kin * Nout + 255) / 256, 256, 0, stream>>>(W, WHt, WLt, Kin, Nout);
    // GEMM
    {
        const int waves  = (N / 16) * (Nout / 64);
        const int blocks = (waves + 7) / 8;
        gemm_wmma_split<<<blocks, 256, 0, stream>>>(XH, XL, WHt, WLt, XP, N, Kin, Nout);
    }
    // X/OUT buffer now dead as input -> reuse as accumulator
    hipMemsetAsync(OUT,  0, (size_t)N * Nout * sizeof(float), stream);
    hipMemsetAsync(mbuf, 0, (size_t)N * H * sizeof(unsigned), stream); // encoded floor
    hipMemsetAsync(den,  0, (size_t)N * H * sizeof(float), stream);

    node_att_kernel<<<(N * H + 255) / 256, 256, 0, stream>>>(XP, att_src, att_dst,
                                                             a_s, a_d, N, H, C, Nout);
    edge_coeff_kernel<<<1, 32, 0, stream>>>(We, att_edge, ce, H, C);

    const int eh_blocks = (E * H + 255) / 256;
    edge_logit_max_kernel<<<eh_blocks, 256, 0, stream>>>(src, dst, ew, a_s, a_d, ce,
                                                         alpha, mbuf, E, H);
    edge_expsum_kernel<<<eh_blocks, 256, 0, stream>>>(dst, alpha, mbuf, den, E, H);
    edge_alpha_kernel<<<eh_blocks, 256, 0, stream>>>(dst, alpha, mbuf, den, E, H);

    const long long ec = (long long)E * Nout;
    edge_scatter_kernel<<<(int)((ec + 255) / 256), 256, 0, stream>>>(src, dst, XP, alpha,
                                                                     OUT, E, Nout, C, H);
    finalize_kernel<<<(N * Nout + 255) / 256, 256, 0, stream>>>(OUT, bias, final_dst,
                                                                N * Nout, Nout, applyElu);
}

extern "C" void kernel_launch(void* const* d_in, const int* in_sizes, int n_in,
                              void* d_out, int out_size, void* d_ws, size_t ws_size,
                              hipStream_t stream) {
    const int D = 64, H = 4, HD = H * D;
    const int N = in_sizes[0] / D;       // 50000
    const int E = in_sizes[2];           // 400000 (edge_weight is [E,1])

    const float* emb = (const float*)d_in[0];
    const int*   ei  = (const int*)d_in[1];
    const int*   src = ei;
    const int*   dst = ei + E;
    const float* ew  = (const float*)d_in[2];

    const float* l0_W  = (const float*)d_in[3];
    const float* l0_as = (const float*)d_in[4];
    const float* l0_ad = (const float*)d_in[5];
    const float* l0_We = (const float*)d_in[6];
    const float* l0_ae = (const float*)d_in[7];
    const float* l0_b  = (const float*)d_in[8];
    const float* l1_W  = (const float*)d_in[9];
    const float* l1_as = (const float*)d_in[10];
    const float* l1_ad = (const float*)d_in[11];
    const float* l1_We = (const float*)d_in[12];
    const float* l1_ae = (const float*)d_in[13];
    const float* l1_b  = (const float*)d_in[14];
    const float* l2_W  = (const float*)d_in[15];
    const float* l2_as = (const float*)d_in[16];
    const float* l2_ad = (const float*)d_in[17];
    const float* l2_We = (const float*)d_in[18];
    const float* l2_ae = (const float*)d_in[19];
    const float* l2_b  = (const float*)d_in[20];

    // workspace carve-up (16B-aligned chunks)
    char* ws = (char*)d_ws;
    size_t off = 0;
    const size_t bigBytes = (size_t)N * HD * sizeof(float);         // 51.2 MB
    float*    bufA  = (float*)(ws + off); off += bigBytes;          // XP
    float*    bufB  = (float*)(ws + off); off += bigBytes;          // X / OUT (ping-pong)
    __bf16*   XH    = (__bf16*)(ws + off); off += (size_t)N * HD * 2;
    __bf16*   XL    = (__bf16*)(ws + off); off += (size_t)N * HD * 2;
    __bf16*   WHt   = (__bf16*)(ws + off); off += (size_t)HD * HD * 2;
    __bf16*   WLt   = (__bf16*)(ws + off); off += (size_t)HD * HD * 2;
    float*    a_s   = (float*)(ws + off); off += (size_t)N * H * sizeof(float);
    float*    a_d   = (float*)(ws + off); off += (size_t)N * H * sizeof(float);
    unsigned* mbuf  = (unsigned*)(ws + off); off += (size_t)N * H * sizeof(unsigned);
    float*    den   = (float*)(ws + off); off += (size_t)N * H * sizeof(float);
    float*    alpha = (float*)(ws + off); off += (size_t)E * H * sizeof(float);
    float*    ce    = (float*)(ws + off); off += 64;
    (void)off; (void)ws_size; (void)n_in; (void)out_size;

    // layer 0: emb[N,64] -> bufB[N,256] (elu applied)
    run_layer(stream, emb, l0_W, l0_as, l0_ad, l0_We, l0_ae, l0_b,
              src, dst, ew, XH, XL, WHt, WLt, bufA, bufB,
              a_s, a_d, mbuf, den, alpha, ce, bufB, N, E, D, HD, H, 1);
    // layer 1: bufB[N,256] -> bufB[N,256] (elu; GEMM inputs packed before reuse)
    run_layer(stream, bufB, l1_W, l1_as, l1_ad, l1_We, l1_ae, l1_b,
              src, dst, ew, XH, XL, WHt, WLt, bufA, bufB,
              a_s, a_d, mbuf, den, alpha, ce, bufB, N, E, HD, HD, H, 1);
    // layer 2: bufB[N,256] -> d_out[N,64] (H=1, no elu)
    run_layer(stream, bufB, l2_W, l2_as, l2_ad, l2_We, l2_ae, l2_b,
              src, dst, ew, XH, XL, WHt, WLt, bufA, bufB,
              a_s, a_d, mbuf, den, alpha, ce, (float*)d_out, N, E, HD, D, 1, 0);
}
